// FusedExperts_88287347736701
// MI455X (gfx1250) — matrix operand order repaired and checked by reading, not compiled, not run on hardware
//
#include <hip/hip_runtime.h>
#include <hip/hip_bf16.h>

// ---------------------------------------------------------------------------
// Fused MoE (FusedExperts) for gfx1250 — expert-grouped bf16 WMMA GEMMs with
// pre-packed bf16 operands and async global->LDS staging (double buffered).
//   per routed pair (t,k), e = idx[t,k]:
//     hg = x_t @ w13[e]; h,g = split(hg); y_t += p[t,k] * (silu(g)*h) @ w2[e]
// ---------------------------------------------------------------------------

typedef __bf16 bf16;
typedef __attribute__((ext_vector_type(16))) __bf16 v16bf;
typedef __attribute__((ext_vector_type(8)))  __bf16 v8bf;
typedef __attribute__((ext_vector_type(8)))  float  v8f;

static constexpr int E      = 8;
static constexpr int TOPK   = 2;
static constexpr int BB     = 2;
static constexpr int S      = 4096;
static constexpr int D      = 1024;       // model dim
static constexpr int H      = 2048;       // ffn dim (w13 is [E, D, 2H])
static constexpr int T      = BB * S;     // 8192 tokens
static constexpr int PAIRS  = T * TOPK;   // 16384 routed pairs
static constexpr int KB     = 32;         // WMMA K per step
static constexpr int BM     = 32;         // token rows per block
static constexpr int BN     = 256;        // output cols per block (8 waves x 4 subtiles)
static constexpr int MAXTILES = (PAIRS + BM - 1) / BM;   // 512 (worst-case skew)

// workspace layout (bytes)
static constexpr size_t WS_CNT    = 0;                       // int[8]
static constexpr size_t WS_CURSOR = 32;                      // int[8]
static constexpr size_t WS_OFFS   = 64;                      // int[8]
static constexpr size_t WS_ZERO   = 128;                     // 128B zero page
static constexpr size_t WS_TOK    = 256;                                   // int[PAIRS]
static constexpr size_t WS_P      = WS_TOK  + (size_t)PAIRS * 4;           // float[PAIRS]
static constexpr size_t WS_XB     = WS_P    + (size_t)PAIRS * 4;           // bf16[T*D]
static constexpr size_t WS_W13T   = WS_XB   + (size_t)T * D * 2;           // bf16[E*2H*D] (transposed)
static constexpr size_t WS_W2T    = WS_W13T + (size_t)E * 2 * H * D * 2;   // bf16[E*D*H]  (transposed)
static constexpr size_t WS_HE     = WS_W2T  + (size_t)E * D * H * 2;       // bf16[PAIRS*H]

// ---------------------------------------------------------------------------
// Async global->LDS staging (CDNA5 ASYNCcnt path).
// The clang builtins for the async copies expect CUDA-language-AS pointer
// types that HIP source cannot spell, so the copies are issued with inline
// asm (ISA 15.18.3: VDST = per-lane LDS byte address, VADDR = 64-bit global
// address, SADDR = off). Completion is enforced with s_wait_asynccnt.
// ---------------------------------------------------------------------------
#if defined(__has_builtin)
#if __has_builtin(__builtin_amdgcn_s_wait_asynccnt)
#define HAVE_ASYNC 1
#endif
#endif
#ifndef HAVE_ASYNC
#define HAVE_ASYNC 0
#endif

__device__ __forceinline__ void copy_b64(const void* g, void* l) {
#if HAVE_ASYNC
  asm volatile("global_load_async_to_lds_b64 %0, %1, off"
               :
               : "v"((uint32_t)(uintptr_t)l), "v"((uint64_t)(uintptr_t)g)
               : "memory");
#else
  *(uint2*)l = *(const uint2*)g;
#endif
}

__device__ __forceinline__ void copy_b128(const void* g, void* l) {
#if HAVE_ASYNC
  asm volatile("global_load_async_to_lds_b128 %0, %1, off"
               :
               : "v"((uint32_t)(uintptr_t)l), "v"((uint64_t)(uintptr_t)g)
               : "memory");
#else
  *(uint4*)l = *(const uint4*)g;
#endif
}

template <int N>
__device__ __forceinline__ void async_wait() {
#if HAVE_ASYNC
  __builtin_amdgcn_s_wait_asynccnt(N);
#endif
}

// ---------------------------------------------------------------------------
// Routing kernels
// ---------------------------------------------------------------------------
__global__ __launch_bounds__(256) void moe_init(float* __restrict__ out,
                                                int* __restrict__ wsi) {
  size_t i = (size_t)blockIdx.x * blockDim.x + threadIdx.x;
  size_t n = (size_t)T * D;
  if (i < n) out[i] = 0.0f;
  if (i < 64) wsi[i] = 0;   // cnt[8], cursor[8], offs[8] (rewritten), zero page
}

__global__ __launch_bounds__(256) void moe_count(const int* __restrict__ idxs,
                                                 int* __restrict__ cnt) {
  int i = blockIdx.x * blockDim.x + threadIdx.x;
  if (i < PAIRS) atomicAdd(&cnt[idxs[i] & (E - 1)], 1);
}

__global__ void moe_scan(const int* __restrict__ cnt, int* __restrict__ offs) {
  if (threadIdx.x == 0) {
    int acc = 0;
    for (int e = 0; e < E; ++e) { offs[e] = acc; acc += cnt[e]; }
  }
}

__global__ __launch_bounds__(256) void moe_scatter(const int* __restrict__ idxs,
                                                   const float* __restrict__ p,
                                                   const int* __restrict__ offs,
                                                   int* __restrict__ cursor,
                                                   int* __restrict__ ptok,
                                                   float* __restrict__ pp) {
  int i = blockIdx.x * blockDim.x + threadIdx.x;
  if (i < PAIRS) {
    int e = idxs[i] & (E - 1);
    int pos = atomicAdd(&cursor[e], 1);
    int slot = offs[e] + pos;
    ptok[slot] = i / TOPK;
    pp[slot] = p[i];
  }
}

// ---------------------------------------------------------------------------
// Packing kernels (run once per launch; streaming, HBM-bound)
// ---------------------------------------------------------------------------
__global__ __launch_bounds__(256) void pack_x(const float* __restrict__ x,
                                              bf16* __restrict__ xb) {
  size_t i = 4 * ((size_t)blockIdx.x * blockDim.x + threadIdx.x);
  float4 v = *(const float4*)(x + i);
  bf16 o[4] = {(bf16)v.x, (bf16)v.y, (bf16)v.z, (bf16)v.w};
  *(uint2*)(xb + i) = *(const uint2*)o;
}

// [batch][R][C] fp32  ->  [batch][C][R] bf16   (32x32 tiles via LDS)
__global__ __launch_bounds__(256) void pack_transpose(const float* __restrict__ src,
                                                      bf16* __restrict__ dst,
                                                      int R, int C) {
  __shared__ bf16 tile[32][33];
  const size_t batch = (size_t)blockIdx.z * R * C;
  const int r0 = blockIdx.y * 32, c0 = blockIdx.x * 32;
  const int tr = threadIdx.x >> 3, tc = (threadIdx.x & 7) * 4;
  float4 v = *(const float4*)(src + batch + (size_t)(r0 + tr) * C + c0 + tc);
  tile[tc + 0][tr] = (bf16)v.x; tile[tc + 1][tr] = (bf16)v.y;
  tile[tc + 2][tr] = (bf16)v.z; tile[tc + 3][tr] = (bf16)v.w;
  __syncthreads();
  bf16* d = dst + batch + (size_t)(c0 + tr) * R + r0 + tc;
  d[0] = tile[tr][tc + 0]; d[1] = tile[tr][tc + 1];
  d[2] = tile[tr][tc + 2]; d[3] = tile[tr][tc + 3];
}

// ---------------------------------------------------------------------------
// WMMA fragment assembly from LDS (two aligned 16B reads -> v16bf)
// ---------------------------------------------------------------------------
__device__ __forceinline__ v16bf frag16(const bf16* lo, const bf16* hi) {
  v8bf a = *(const v8bf*)lo;
  v8bf b = *(const v8bf*)hi;
  v16bf r;
#pragma unroll
  for (int i = 0; i < 8; ++i) { r[i] = a[i]; r[i + 8] = b[i]; }
  return r;
}

__device__ __forceinline__ v8f wmma_bf16(v16bf a, v16bf b, v8f c) {
  return __builtin_amdgcn_wmma_f32_16x16x32_bf16(false, a, false, b,
                                                 (short)0, c, false, false);
}

// ---------------------------------------------------------------------------
// GEMM1: hg = Xg @ w13[e]; he = silu(g) * h -> bf16 workspace
// grid: (MAXTILES, H/BN, E), block 256. Wave grid 2(M)x4(N); wave = 16x64 per matrix.
// ---------------------------------------------------------------------------
__global__ __launch_bounds__(256) void moe_gemm1(
    const bf16* __restrict__ xb, const bf16* __restrict__ w13t,
    const int* __restrict__ ptok, const int* __restrict__ cnt,
    const int* __restrict__ offs, const bf16* __restrict__ zp,
    bf16* __restrict__ he) {
  const int e = blockIdx.z;
  const int count = cnt[e];
  const int tile = blockIdx.x;
  if (tile * BM >= count) return;
  const int n0 = blockIdx.y * BN;          // h-column start in [0, H)
  const int base = offs[e] + tile * BM;

  __shared__ __align__(16) bf16 As[2][BM][KB];   // 2 x 2KB
  __shared__ __align__(16) bf16 Bh[2][BN][KB];   // 2 x 16KB  (transposed [n][k])
  __shared__ __align__(16) bf16 Bg[2][BN][KB];   // 2 x 16KB

  const int tid = threadIdx.x;
  const int lane = tid & 31, wave = tid >> 5;
  const int wm = wave >> 2, wn = wave & 3;
  const int hf = lane >> 4, l16 = lane & 15;

  // A staging: 32 rows x 64B, one b64 per thread
  const int arow = tid >> 3, ake = (tid & 7) * 4;    // element offset in row chunk
  const bool avalid = (tile * BM + arow) < count;
  const bf16* aptr = zp + ake;
  if (avalid) aptr = xb + (size_t)ptok[base + arow] * D + ake;
  // B staging: one row (64B) per thread, 4 x b128
  const bf16* bhp = w13t + ((size_t)e * 2 * H + n0 + tid) * D;
  const bf16* bgp = w13t + ((size_t)e * 2 * H + H + n0 + tid) * D;

  v8f acch[4], accg[4];
  const v8f zero = {0.f, 0.f, 0.f, 0.f, 0.f, 0.f, 0.f, 0.f};
#pragma unroll
  for (int t = 0; t < 4; ++t) { acch[t] = zero; accg[t] = zero; }

  auto stage = [&](int buf, int k0) {
    copy_b64(avalid ? (aptr + k0) : aptr, &As[buf][arow][ake]);
#pragma unroll
    for (int j = 0; j < 4; ++j) {
      copy_b128(bhp + k0 + j * 8, &Bh[buf][tid][j * 8]);
      copy_b128(bgp + k0 + j * 8, &Bg[buf][tid][j * 8]);
    }
  };

  stage(0, 0);
  int buf = 0;
  for (int k0 = 0; k0 < D; k0 += KB, buf ^= 1) {
    const bool more = (k0 + KB) < D;
    if (more) { stage(buf ^ 1, k0 + KB); async_wait<9>(); }
    else      { async_wait<0>(); }
    __syncthreads();

    // A frag: lane row m, K = {8*hf + j, 16 + 8*hf + j}
    v16bf af = frag16(&As[buf][wm * 16 + l16][hf * 8],
                      &As[buf][wm * 16 + l16][16 + hf * 8]);
#pragma unroll
    for (int t = 0; t < 4; ++t) {
      const int n = wn * 64 + t * 16 + l16;
      v16bf bh = frag16(&Bh[buf][n][hf * 8], &Bh[buf][n][16 + hf * 8]);
      acch[t] = wmma_bf16(af, bh, acch[t]);
      v16bf bg = frag16(&Bg[buf][n][hf * 8], &Bg[buf][n][16 + hf * 8]);
      accg[t] = wmma_bf16(af, bg, accg[t]);
    }
    __syncthreads();
  }

  // epilogue: he = silu(g) * h. C/D layout: row = 8*hf + r, col = l16.
#pragma unroll
  for (int t = 0; t < 4; ++t) {
#pragma unroll
    for (int r = 0; r < 8; ++r) {
      const int m = wm * 16 + hf * 8 + r;
      if (tile * BM + m < count) {
        float gv = accg[t][r];
        float sv = gv / (1.0f + __expf(-gv));
        he[(size_t)(base + m) * H + n0 + wn * 64 + t * 16 + l16] =
            (bf16)(acch[t][r] * sv);
      }
    }
  }
}

// ---------------------------------------------------------------------------
// GEMM2: y[token] += p * (he @ w2[e]); f32 atomics into out
// grid: (MAXTILES, D/BN, E), block 256
// ---------------------------------------------------------------------------
__global__ __launch_bounds__(256) void moe_gemm2(
    const bf16* __restrict__ he, const bf16* __restrict__ w2t,
    const int* __restrict__ ptok, const float* __restrict__ pp,
    const int* __restrict__ cnt, const int* __restrict__ offs,
    const bf16* __restrict__ zp, float* __restrict__ out) {
  const int e = blockIdx.z;
  const int count = cnt[e];
  const int tile = blockIdx.x;
  if (tile * BM >= count) return;
  const int n0 = blockIdx.y * BN;
  const int base = offs[e] + tile * BM;

  __shared__ __align__(16) bf16 As[2][BM][KB];
  __shared__ __align__(16) bf16 Bs[2][BN][KB];

  const int tid = threadIdx.x;
  const int lane = tid & 31, wave = tid >> 5;
  const int wm = wave >> 2, wn = wave & 3;
  const int hf = lane >> 4, l16 = lane & 15;

  const int arow = tid >> 3, ake = (tid & 7) * 4;
  const bool avalid = (tile * BM + arow) < count;
  const bf16* aptr = zp + ake;
  if (avalid) aptr = he + (size_t)(base + arow) * H + ake;
  const bf16* bp = w2t + ((size_t)e * D + n0 + tid) * H;

  v8f acc[4];
  const v8f zero = {0.f, 0.f, 0.f, 0.f, 0.f, 0.f, 0.f, 0.f};
#pragma unroll
  for (int t = 0; t < 4; ++t) acc[t] = zero;

  auto stage = [&](int buf, int k0) {
    copy_b64(avalid ? (aptr + k0) : aptr, &As[buf][arow][ake]);
#pragma unroll
    for (int j = 0; j < 4; ++j)
      copy_b128(bp + k0 + j * 8, &Bs[buf][tid][j * 8]);
  };

  stage(0, 0);
  int buf = 0;
  for (int k0 = 0; k0 < H; k0 += KB, buf ^= 1) {
    const bool more = (k0 + KB) < H;
    if (more) { stage(buf ^ 1, k0 + KB); async_wait<5>(); }
    else      { async_wait<0>(); }
    __syncthreads();

    v16bf af = frag16(&As[buf][wm * 16 + l16][hf * 8],
                      &As[buf][wm * 16 + l16][16 + hf * 8]);
#pragma unroll
    for (int t = 0; t < 4; ++t) {
      const int n = wn * 64 + t * 16 + l16;
      v16bf bb = frag16(&Bs[buf][n][hf * 8], &Bs[buf][n][16 + hf * 8]);
      acc[t] = wmma_bf16(af, bb, acc[t]);
    }
    __syncthreads();
  }

#pragma unroll
  for (int t = 0; t < 4; ++t) {
#pragma unroll
    for (int r = 0; r < 8; ++r) {
      const int m = wm * 16 + hf * 8 + r;
      if (tile * BM + m < count) {
        const int slot = base + m;
        atomicAdd(out + (size_t)ptok[slot] * D + n0 + wn * 64 + t * 16 + l16,
                  pp[slot] * acc[t][r]);
      }
    }
  }
}

// ---------------------------------------------------------------------------
// Launch
// ---------------------------------------------------------------------------
extern "C" void kernel_launch(void* const* d_in, const int* in_sizes, int n_in,
                              void* d_out, int out_size, void* d_ws, size_t ws_size,
                              hipStream_t stream) {
  (void)in_sizes; (void)n_in; (void)out_size; (void)ws_size;
  const float* x   = (const float*)d_in[0];   // [T, D]
  const float* ep  = (const float*)d_in[1];   // [T, TOPK]
  const int*   ei  = (const int*)d_in[2];     // [T, TOPK]
  const float* w13 = (const float*)d_in[3];   // [E, D, 2H]
  const float* w2  = (const float*)d_in[4];   // [E, H, D]
  float* out = (float*)d_out;                 // [T, D]

  char* ws = (char*)d_ws;
  int*   cnt    = (int*)(ws + WS_CNT);
  int*   cursor = (int*)(ws + WS_CURSOR);
  int*   offs   = (int*)(ws + WS_OFFS);
  bf16*  zp     = (bf16*)(ws + WS_ZERO);
  int*   ptok   = (int*)(ws + WS_TOK);
  float* pp     = (float*)(ws + WS_P);
  bf16*  xb     = (bf16*)(ws + WS_XB);
  bf16*  w13t   = (bf16*)(ws + WS_W13T);
  bf16*  w2t    = (bf16*)(ws + WS_W2T);
  bf16*  he     = (bf16*)(ws + WS_HE);

  const size_t nOut = (size_t)T * D;
  moe_init<<<(unsigned)((nOut + 255) / 256), 256, 0, stream>>>(out, (int*)ws);
  moe_count<<<(PAIRS + 255) / 256, 256, 0, stream>>>(ei, cnt);
  moe_scan<<<1, 32, 0, stream>>>(cnt, offs);
  moe_scatter<<<(PAIRS + 255) / 256, 256, 0, stream>>>(ei, ep, offs, cursor, ptok, pp);

  pack_x<<<(unsigned)(nOut / 4 / 256), 256, 0, stream>>>(x, xb);
  {
    dim3 g(2 * H / 32, D / 32, E);   // (128, 32, 8)
    pack_transpose<<<g, 256, 0, stream>>>(w13, w13t, D, 2 * H);
  }
  {
    dim3 g(D / 32, H / 32, E);       // (32, 64, 8)
    pack_transpose<<<g, 256, 0, stream>>>(w2, w2t, H, D);
  }

  dim3 g1(MAXTILES, H / BN, E);      // (512, 8, 8)
  moe_gemm1<<<g1, 256, 0, stream>>>(xb, w13t, ptok, cnt, offs, zp, he);
  dim3 g2(MAXTILES, D / BN, E);      // (512, 4, 8)
  moe_gemm2<<<g2, 256, 0, stream>>>(he, w2t, ptok, pp, cnt, offs, zp, out);
}
